// SelfAttention_Sparse_27504970564329
// MI455X (gfx1250) — matrix operand order repaired and checked by reading, not compiled, not run on hardware
//
#include <hip/hip_runtime.h>
#include <cstdint>
#include <cstddef>

// Problem constants (from reference): N=4, S=4096, E=1024, H=16, HD=64, BS=256
#define N_B 4
#define SEQ 4096
#define EMB 1024
#define NH  16
#define HD  64
#define BSZ 256
#define NA  (SEQ / BSZ)   // 16

typedef __bf16 bf16_t;
typedef __attribute__((ext_vector_type(16))) __bf16 bf16x16;
typedef __attribute__((ext_vector_type(8)))  __bf16 bf16x8;
typedef __attribute__((ext_vector_type(8)))  float  f32x8;
typedef __attribute__((ext_vector_type(4)))  unsigned int u32x4;
typedef __attribute__((ext_vector_type(8)))  int i32x8;
typedef __attribute__((ext_vector_type(4)))  int i32x4;

union ABFrag { bf16x16 v; bf16x8 h[2]; };

__device__ __forceinline__ f32x8 wmma_bf16(const bf16x16& a, const bf16x16& b, const f32x8& c) {
  // D = A(16x32 bf16) * B(32x16 bf16) + C(16x16 f32)
  return __builtin_amdgcn_wmma_f32_16x16x32_bf16(false, a, false, b, (short)0, c, false, false);
}

// ---------------------------------------------------------------------------
// TDM: async-load a 2-D bf16 tile [rows x cols] (row stride strideElems) from
// global memory into LDS at byte offset ldsOff. Issued per-wave (EXEC ignored);
// caller waits with s_wait_tensorcnt then a workgroup barrier.
// Descriptor layout per CDNA5 ISA ch. 8 (D# groups 0/1; groups 2/3 zero for 2D).
// ---------------------------------------------------------------------------
__device__ __forceinline__ void tdm_load_tile_bf16(const bf16_t* gsrc, unsigned ldsOff,
                                                   unsigned rows, unsigned cols,
                                                   unsigned strideElems) {
  unsigned long long ga = (unsigned long long)(uintptr_t)gsrc;
  u32x4 g0;
  g0[0] = 1u;                                  // count=1, is_restore=0, gather=0
  g0[1] = ldsOff;                              // lds_addr (bytes)
  g0[2] = (unsigned)ga;                        // global_addr[31:0]
  g0[3] = (unsigned)((ga >> 32) & 0x1FFFFFFu) | (2u << 30);  // addr[56:32] | type=2
  i32x8 g1;
  g1[0] = 0x00010000;                          // workgroup_mask=0, data_size=1 (2B)
  g1[1] = (int)(cols << 16);                   // tensor_dim0[15:0]
  g1[2] = (int)(rows << 16);                   // tensor_dim0 hi=0, tensor_dim1[15:0]
  g1[3] = (int)(cols << 16);                   // tensor_dim1 hi=0, tile_dim0
  g1[4] = (int)rows;                           // tile_dim1 (tile_dim2=0)
  g1[5] = (int)strideElems;                    // tensor_dim0_stride[31:0]
  g1[6] = 0;                                   // stride hi, dim1_stride lo
  g1[7] = 0;
  i32x4 gz = {0, 0, 0, 0};
#if __clang_major__ >= 23
  i32x8 gz8 = {0, 0, 0, 0, 0, 0, 0, 0};
  __builtin_amdgcn_tensor_load_to_lds(g0, g1, gz, gz, gz8, 0);
#else
  __builtin_amdgcn_tensor_load_to_lds(g0, g1, gz, gz, 0);
#endif
}

// ---------------------------------------------------------------------------
// f32 -> bf16 elementwise conversion (weights)
// ---------------------------------------------------------------------------
__global__ void __launch_bounds__(256)
cvt_f32_to_bf16(const float* __restrict__ in, bf16_t* __restrict__ out, int n) {
  int i = blockIdx.x * blockDim.x + threadIdx.x;
  int stride = gridDim.x * blockDim.x;
  for (; i < n; i += stride) out[i] = (bf16_t)in[i];
}

// ---------------------------------------------------------------------------
// GEMM: out[M,Nout] = X[M,Kd] @ W[Nout,Kd]^T + bias   (x @ W.T + b)
// 256 threads = 8 waves; WG tile 256(M) x 64(N); wave owns 32 rows; K-step 32.
// 8 accumulating WMMAs per wave per K-step between barriers.
// InT = float (convert to bf16 while staging) or bf16 (direct copy).
// ---------------------------------------------------------------------------
template <typename InT, typename OutT>
__global__ void __launch_bounds__(256)
gemm_bias_wmma(const InT* __restrict__ X, const bf16_t* __restrict__ W,
               const float* __restrict__ bias, OutT* __restrict__ out,
               int M, int Kd, int Nout) {
  __shared__ __attribute__((aligned(32))) bf16_t Alds[256 * 32];  // 16 KB
  __shared__ __attribute__((aligned(32))) bf16_t Blds[64 * 32];   //  4 KB

  const int tid  = threadIdx.x;
  const int wave = tid >> 5;
  const int lane = tid & 31;
  const int m    = lane & 15;
  const int hi   = lane >> 4;
  const int m0   = blockIdx.y * 256;
  const int n0   = blockIdx.x * 64;

  f32x8 acc[2][4] = {};

  for (int k0 = 0; k0 < Kd; k0 += 32) {
    // ---- stage A tile 256x32 (bf16): 1024 groups of 8 elements ----
#pragma unroll
    for (int i = 0; i < 4; ++i) {
      int g   = tid + i * 256;
      int row = g >> 2;
      int c8  = (g & 3) * 8;
      if constexpr (sizeof(InT) == 4) {
        const float* src = (const float*)X + (size_t)(m0 + row) * Kd + k0 + c8;
        float4 f0 = *(const float4*)src;
        float4 f1 = *(const float4*)(src + 4);
        bf16x8 o;
        o[0] = (bf16_t)f0.x; o[1] = (bf16_t)f0.y; o[2] = (bf16_t)f0.z; o[3] = (bf16_t)f0.w;
        o[4] = (bf16_t)f1.x; o[5] = (bf16_t)f1.y; o[6] = (bf16_t)f1.z; o[7] = (bf16_t)f1.w;
        *(bf16x8*)&Alds[row * 32 + c8] = o;
        if (k0 + 32 < Kd) __builtin_prefetch(src + 32, 0, 1);
      } else {
        const bf16_t* src = (const bf16_t*)X + (size_t)(m0 + row) * Kd + k0 + c8;
        *(uint4*)&Alds[row * 32 + c8] = *(const uint4*)src;
        if (k0 + 32 < Kd) __builtin_prefetch(src + 32, 0, 1);
      }
    }
    // ---- stage B tile 64x32 from W rows (W[n][k] is already B column n) ----
    {
      int row = tid >> 2;
      int c8  = (tid & 3) * 8;
      *(uint4*)&Blds[row * 32 + c8] =
          *(const uint4*)(W + (size_t)(n0 + row) * Kd + k0 + c8);
    }
    __syncthreads();

    // A fragments: rows wave*32 + sub*16 + m; halves at kOff=hi*16 and hi*16+8
    ABFrag a[2];
#pragma unroll
    for (int sub = 0; sub < 2; ++sub) {
      const bf16_t* arow = &Alds[(wave * 32 + sub * 16 + m) * 32];
      a[sub].h[0] = *(const bf16x8*)(arow + hi * 16);
      a[sub].h[1] = *(const bf16x8*)(arow + hi * 16 + 8);
    }
#pragma unroll
    for (int t = 0; t < 4; ++t) {
      bf16x16 b = *(const bf16x16*)&Blds[(t * 16 + m) * 32 + hi * 16];
      acc[0][t] = wmma_bf16(a[0].v, b, acc[0][t]);
      acc[1][t] = wmma_bf16(a[1].v, b, acc[1][t]);
    }
    __syncthreads();
  }

  // epilogue: bias + store (C layout: row = r + 8*hi, col = m per 16-tile)
#pragma unroll
  for (int sub = 0; sub < 2; ++sub) {
#pragma unroll
    for (int t = 0; t < 4; ++t) {
      int col = n0 + t * 16 + m;
      float bcol = bias[col];
#pragma unroll
      for (int r = 0; r < 8; ++r) {
        int row = m0 + wave * 32 + sub * 16 + r + 8 * hi;
        float v = acc[sub][t][r] + bcol;
        if constexpr (sizeof(OutT) == 4)
          ((float*)out)[(size_t)row * Nout + col] = v;
        else
          ((bf16_t*)out)[(size_t)row * Nout + col] = (bf16_t)v;
      }
    }
  }
}

// ---------------------------------------------------------------------------
// Block-local attention, flash-style online softmax.
// Grid: (BSZ/128, NA, N_B*NH). 256 threads = 8 waves; wave owns 16 q rows.
// LDS: K block [256][64] bf16 (32KB, loaded by TDM) + V^T [64][256] bf16 (32KB)
//      + per-wave P staging [8][16][32] bf16 (8KB) = 72KB dynamic.
// ---------------------------------------------------------------------------
__global__ void __launch_bounds__(256)
block_attn_wmma(const bf16_t* __restrict__ Q, const bf16_t* __restrict__ K,
                const bf16_t* __restrict__ V, bf16_t* __restrict__ O,
                const unsigned char* __restrict__ mask) {
  extern __shared__ __attribute__((aligned(32))) char smem[];
  bf16_t* Klds = (bf16_t*)smem;            // [BSZ][HD]   at LDS offset 0
  bf16_t* VT   = Klds + BSZ * HD;          // [HD][BSZ]
  bf16_t* Plds = VT + HD * BSZ;            // [8 waves][16][32]

  const int tid   = threadIdx.x;
  const int wave  = tid >> 5;
  const int lane  = tid & 31;
  const int m     = lane & 15;
  const int hi    = lane >> 4;
  const int qhalf = blockIdx.x;            // 0..1
  const int a     = blockIdx.y;            // 0..15
  const int nh    = blockIdx.z;            // 0..63
  const int n     = nh >> 4;
  const int h     = nh & 15;
  const size_t base = ((size_t)n * SEQ + (size_t)a * BSZ) * EMB + (size_t)h * HD;

  // ---- stage K block [256][64] via Tensor Data Mover (wave 0 issues) ----
  if (wave == 0) {
    tdm_load_tile_bf16(K + base, /*ldsOff=*/0, /*rows=*/BSZ, /*cols=*/HD,
                       /*strideElems=*/EMB);
    __builtin_amdgcn_s_wait_tensorcnt(0);
  }
  // ---- stage V transposed VT[d][k] (manual: TDM cannot transpose) ----
#pragma unroll
  for (int i = 0; i < 32; ++i) {
    int g  = tid + i * 256;                // 8192 (k, d-pair) slots
    int k  = g >> 5;
    int d2 = (g & 31) * 2;
    const bf16_t* vr = V + base + (size_t)k * EMB;
    VT[(d2    ) * BSZ + k] = vr[d2];
    VT[(d2 + 1) * BSZ + k] = vr[d2 + 1];
  }
  __syncthreads();

  const float scale   = 0.125f;                       // 1/sqrt(64)
  const float maskAdd = mask[0] ? 0.0f : -1.25e19f;   // (-1e20)/sqrt(64)

  // ---- Q fragments for this wave's 16 rows, 2 d-chunks of 32 ----
  ABFrag aq[2];
  const bf16_t* qrow = Q + base + (size_t)(qhalf * 128 + wave * 16 + m) * EMB;
#pragma unroll
  for (int c = 0; c < 2; ++c) {
    aq[c].h[0] = *(const bf16x8*)(qrow + c * 32 + hi * 16);
    aq[c].h[1] = *(const bf16x8*)(qrow + c * 32 + hi * 16 + 8);
  }

  f32x8 acc[4] = {};
  float rmax[8], rsum[8];
#pragma unroll
  for (int r = 0; r < 8; ++r) { rmax[r] = -3.0e38f; rsum[r] = 0.0f; }

  bf16_t* Pw = Plds + wave * 16 * 32;      // wave-private P staging [16][32]

  for (int kc = 0; kc < BSZ; kc += 32) {
    // ---- scores: S = Q @ K^T, two 16-wide column tiles ----
    f32x8 s[2] = {};
#pragma unroll
    for (int c = 0; c < 2; ++c) {
#pragma unroll
      for (int t = 0; t < 2; ++t) {
        bf16x16 bk = *(const bf16x16*)&Klds[(kc + t * 16 + m) * HD + c * 32 + hi * 16];
        s[t] = wmma_bf16(aq[c].v, bk, s[t]);
      }
    }
    // ---- online softmax over this 32-wide chunk ----
#pragma unroll
    for (int r = 0; r < 8; ++r) {
      float v0 = s[0][r] * scale + maskAdd;
      float v1 = s[1][r] * scale + maskAdd;
      float mx = fmaxf(v0, v1);
#pragma unroll
      for (int off = 1; off < 16; off <<= 1)
        mx = fmaxf(mx, __shfl_xor(mx, off, 32));
      float mnew = fmaxf(rmax[r], mx);
      float corr = __expf(rmax[r] - mnew);
      rmax[r] = mnew;
      float p0 = __expf(v0 - mnew);
      float p1 = __expf(v1 - mnew);
      float ls = p0 + p1;
#pragma unroll
      for (int off = 1; off < 16; off <<= 1)
        ls += __shfl_xor(ls, off, 32);
      rsum[r] = rsum[r] * corr + ls;
#pragma unroll
      for (int t = 0; t < 4; ++t) acc[t][r] *= corr;
      // write P in row-major [16][32] (C-layout -> LDS -> A-fragment re-layout)
      Pw[(r + 8 * hi) * 32 + m]      = (bf16_t)p0;
      Pw[(r + 8 * hi) * 32 + 16 + m] = (bf16_t)p1;
    }
    // ---- A fragment of P (wave-private LDS, DS-ordered within wave) ----
    ABFrag ap;
    ap.h[0] = *(const bf16x8*)&Pw[m * 32 + hi * 16];
    ap.h[1] = *(const bf16x8*)&Pw[m * 32 + hi * 16 + 8];
    // ---- O += P @ V over 4 d-tiles of 16 ----
#pragma unroll
    for (int t = 0; t < 4; ++t) {
      bf16x16 bv = *(const bf16x16*)&VT[(t * 16 + m) * BSZ + kc + hi * 16];
      acc[t] = wmma_bf16(ap.v, bv, acc[t]);
    }
  }

  // ---- normalize and write context [n, s, e] bf16 ----
#pragma unroll
  for (int r = 0; r < 8; ++r) {
    float inv = 1.0f / rsum[r];
    size_t row = base + (size_t)(qhalf * 128 + wave * 16 + r + 8 * hi) * EMB;
#pragma unroll
    for (int t = 0; t < 4; ++t)
      O[row + t * 16 + m] = (bf16_t)(acc[t][r] * inv);
  }
}

// ---------------------------------------------------------------------------
extern "C" void kernel_launch(void* const* d_in, const int* in_sizes, int n_in,
                              void* d_out, int out_size, void* d_ws, size_t ws_size,
                              hipStream_t stream) {
  (void)in_sizes; (void)n_in; (void)out_size; (void)ws_size;

  const float* values = (const float*)d_in[0];
  const float* keys   = (const float*)d_in[1];
  const float* query  = (const float*)d_in[2];
  const unsigned char* mask = (const unsigned char*)d_in[3];
  // d_in[4] = block_size (== BSZ, compile-time)
  const float* Wv = (const float*)d_in[5];
  const float* bv = (const float*)d_in[6];
  const float* Wk = (const float*)d_in[7];
  const float* bk = (const float*)d_in[8];
  const float* Wq = (const float*)d_in[9];
  const float* bq = (const float*)d_in[10];
  const float* Wo = (const float*)d_in[11];
  const float* bo = (const float*)d_in[12];

  const size_t NSE = (size_t)N_B * SEQ * EMB;  // 16,777,216 elems
  const size_t EE  = (size_t)EMB * EMB;        //  1,048,576 elems

  // Workspace layout (~142.6 MB total, all regions fully rewritten every call):
  char* ws = (char*)d_ws;
  size_t off = 0;
  bf16_t* qb  = (bf16_t*)(ws + off); off += NSE * sizeof(bf16_t);
  bf16_t* kb  = (bf16_t*)(ws + off); off += NSE * sizeof(bf16_t);
  bf16_t* vb  = (bf16_t*)(ws + off); off += NSE * sizeof(bf16_t);
  bf16_t* ctx = (bf16_t*)(ws + off); off += NSE * sizeof(bf16_t);
  bf16_t* Wqb = (bf16_t*)(ws + off); off += EE * sizeof(bf16_t);
  bf16_t* Wkb = (bf16_t*)(ws + off); off += EE * sizeof(bf16_t);
  bf16_t* Wvb = (bf16_t*)(ws + off); off += EE * sizeof(bf16_t);
  bf16_t* Wob = (bf16_t*)(ws + off); off += EE * sizeof(bf16_t);

  // 1) convert weights to bf16
  cvt_f32_to_bf16<<<1024, 256, 0, stream>>>(Wq, Wqb, (int)EE);
  cvt_f32_to_bf16<<<1024, 256, 0, stream>>>(Wk, Wkb, (int)EE);
  cvt_f32_to_bf16<<<1024, 256, 0, stream>>>(Wv, Wvb, (int)EE);
  cvt_f32_to_bf16<<<1024, 256, 0, stream>>>(Wo, Wob, (int)EE);

  // 2) QKV projections (f32 in, bf16 out)
  dim3 ggrid(EMB / 64, (N_B * SEQ) / 256);   // (16, 64)
  gemm_bias_wmma<float, bf16_t><<<ggrid, 256, 0, stream>>>(
      query,  Wqb, bq, qb, N_B * SEQ, EMB, EMB);
  gemm_bias_wmma<float, bf16_t><<<ggrid, 256, 0, stream>>>(
      keys,   Wkb, bk, kb, N_B * SEQ, EMB, EMB);
  gemm_bias_wmma<float, bf16_t><<<ggrid, 256, 0, stream>>>(
      values, Wvb, bv, vb, N_B * SEQ, EMB, EMB);

  // 3) block-local attention (TDM K-staging + WMMA flash loop)
  dim3 agrid(BSZ / 128, NA, N_B * NH);       // (2, 16, 64)
  size_t shmem = (size_t)(BSZ * HD + HD * BSZ + 8 * 16 * 32) * sizeof(bf16_t); // 73728
  block_attn_wmma<<<agrid, 256, shmem, stream>>>(qb, kb, vb, ctx, mask);

  // 4) output projection (bf16 in, f32 out -> d_out)
  gemm_bias_wmma<bf16_t, float><<<ggrid, 256, 0, stream>>>(
      ctx, Wob, bo, (float*)d_out, N_B * SEQ, EMB, EMB);
}
